// TensorizedSquareModule_39109972197963
// MI455X (gfx1250) — compile-verified, matrix operand order checked by reading, not run on hardware
//
#include <hip/hip_runtime.h>
#include <hip/hip_bf16.h>
#include <stdint.h>

// ---------------------------------------------------------------------------
// Problem sizes (fixed by the reference)
// ---------------------------------------------------------------------------
#define B_SZ   8
#define S_SZ   2048
#define IN_F   2048
#define OUT_F  2048
#define MTOT   (B_SZ * S_SZ)     // 16384

// GEMM tiling
#define BM 128
#define BN 128
#define BK 64
#define LDT 72                    // padded LDS row stride in bf16 elems (144 B)

typedef __attribute__((ext_vector_type(16))) __bf16 bf16x16;
typedef __attribute__((ext_vector_type(8)))  __bf16 bf16x8;
typedef __attribute__((ext_vector_type(8)))  float  v8f;
typedef __attribute__((ext_vector_type(4)))  int    i32x4;

union FragU { bf16x16 v; bf16x8 h[2]; };

// CDNA5 async global->LDS copy path (ASYNCcnt), guarded: falls back to the
// proven synchronous VGPR staging path if the builtins are absent.
#if defined(__gfx1250__) && __has_builtin(__builtin_amdgcn_global_load_async_to_lds_b128)
#define USE_ASYNC 1
#else
#define USE_ASYNC 0
#endif

#if USE_ASYNC
__device__ __forceinline__ void async_copy16(const void* g, void* l) {
  __builtin_amdgcn_global_load_async_to_lds_b128(
      (__attribute__((address_space(1))) i32x4*)(g),
      (__attribute__((address_space(3))) i32x4*)(l),
      /*offset=*/0, /*cpol=*/0);
}
__device__ __forceinline__ void wait_asynccnt_8() {
#if __has_builtin(__builtin_amdgcn_s_wait_asynccnt)
  __builtin_amdgcn_s_wait_asynccnt(8);
#else
  asm volatile("s_wait_asynccnt 0x8" ::: "memory");
#endif
  asm volatile("" ::: "memory");
}
__device__ __forceinline__ void wait_asynccnt_0() {
#if __has_builtin(__builtin_amdgcn_s_wait_asynccnt)
  __builtin_amdgcn_s_wait_asynccnt(0);
#else
  asm volatile("s_wait_asynccnt 0x0" ::: "memory");
#endif
  asm volatile("" ::: "memory");
}
#endif

__device__ __forceinline__ unsigned short f32_to_bf16_rne(float f) {
  unsigned int u = __float_as_uint(f);
  unsigned int r = u + 0x7fffu + ((u >> 16) & 1u);
  return (unsigned short)(r >> 16);
}

// ---------------------------------------------------------------------------
// Kernel 1: Wb[o][i] = bf16( weight[o][i] + sum_r L0[r][a0][b0]*L1[r][a1][b1]*L2[r][a2][b2] )
// o = a0*169 + a1*13 + a2 ; i = b0*169 + b1*13 + b2  (13^3 = 2197 >= 2048)
// leaf layout: [order=3][rank=4][13][13]
// ---------------------------------------------------------------------------
__global__ void build_w_kernel(const float* __restrict__ leaf,
                               const float* __restrict__ weight,
                               unsigned short* __restrict__ Wb) {
  int idx = blockIdx.x * blockDim.x + threadIdx.x;   // over 2048*2048
  int o = idx >> 11;
  int i = idx & 2047;
  int a0 = o / 169; int ot = o - a0 * 169; int a1 = ot / 13; int a2 = ot - a1 * 13;
  int b0 = i / 169; int it = i - b0 * 169; int b1 = it / 13; int b2 = it - b1 * 13;
  float delta = 0.f;
#pragma unroll
  for (int r = 0; r < 4; ++r) {
    float l0 = leaf[((0 * 4 + r) * 13 + a0) * 13 + b0];
    float l1 = leaf[((1 * 4 + r) * 13 + a1) * 13 + b1];
    float l2 = leaf[((2 * 4 + r) * 13 + a2) * 13 + b2];
    delta += l0 * l1 * l2;
  }
  float v = weight[idx] + delta;
  Wb[idx] = f32_to_bf16_rne(v);
}

// ---------------------------------------------------------------------------
// Kernel 2: f32 -> bf16 conversion of the activations (float4 per thread)
// ---------------------------------------------------------------------------
__global__ void cvt_x_kernel(const float* __restrict__ x,
                             unsigned short* __restrict__ xb) {
  int idx = blockIdx.x * blockDim.x + threadIdx.x;   // one float4 each
  float4 v = ((const float4*)x)[idx];
  unsigned int lo = (unsigned int)f32_to_bf16_rne(v.x) |
                    ((unsigned int)f32_to_bf16_rne(v.y) << 16);
  unsigned int hi = (unsigned int)f32_to_bf16_rne(v.z) |
                    ((unsigned int)f32_to_bf16_rne(v.w) << 16);
  ((uint2*)xb)[idx] = make_uint2(lo, hi);
}

// ---------------------------------------------------------------------------
// Kernel 3: C[m][n] = sum_k Ab[m][k] * Wb[n][k] + bias[n]
// Ab: [MTOT][IN_F] bf16 row-major ; Wb: [OUT_F][IN_F] bf16 row-major
// Block tile 128x128, BK=64; 8 waves (wave32) in 2(M)x4(N) grid;
// wave tile 64x32 = 4x2 WMMA 16x16 accumulators.
// Double-buffered LDS filled by async global->LDS copies (ASYNCcnt).
// ---------------------------------------------------------------------------
__global__ __launch_bounds__(256, 2) void gemm_kernel(
    const unsigned short* __restrict__ Ab,
    const unsigned short* __restrict__ Wb,
    const float* __restrict__ bias,
    float* __restrict__ C) {
  __shared__ __align__(16) unsigned short sA[2][BM * LDT];
  __shared__ __align__(16) unsigned short sB[2][BN * LDT];

  const int tid   = threadIdx.x;
  const int lane  = tid & 31;
  const int wid   = tid >> 5;
  const int waveM = wid >> 2;          // 0..1  -> 64 rows of M each
  const int waveN = wid & 3;           // 0..3  -> 32 cols of N each
  const int bm = blockIdx.y * BM;
  const int bn = blockIdx.x * BN;

  v8f acc[4][2];
#pragma unroll
  for (int sm = 0; sm < 4; ++sm)
#pragma unroll
    for (int sn = 0; sn < 2; ++sn) {
      v8f z = {0.f, 0.f, 0.f, 0.f, 0.f, 0.f, 0.f, 0.f};
      acc[sm][sn] = z;
    }

  // Per-lane fragment geometry (16-bit A/B VGPR layout, ISA 7.12.2):
  // lanes 0-15 hold row (lane), K-base 0; lanes 16-31 hold row (lane-16), K-base 8.
  // Fragment = two contiguous 16B chunks: K in [kb, kb+8) and [kb+16, kb+24).
  const int m_lane = lane & 15;
  const int kb     = (lane >> 4) << 3;   // 0 or 8

  const int nTiles = IN_F / BK;          // 32

#if USE_ASYNC
  // Each thread owns 4 A-chunks + 4 B-chunks (16B each) per K-tile => 8 async
  // loads per thread per tile. Per-wave async loads complete in order, so
  // s_wait_asynccnt <= 8 retires the previous tile's batch.
  auto issue_tile = [&](int buf, int k0) {
#pragma unroll
    for (int c = 0; c < 4; ++c) {
      int cid = tid + c * 256;           // 0..1023
      int row = cid >> 3;                // 0..127
      int ch  = (cid & 7) << 3;          // element offset within row (0..56)
      async_copy16(&Ab[(size_t)(bm + row) * IN_F + k0 + ch],
                   &sA[buf][row * LDT + ch]);
      async_copy16(&Wb[(size_t)(bn + row) * IN_F + k0 + ch],
                   &sB[buf][row * LDT + ch]);
    }
  };

  issue_tile(0, 0);

  for (int kt = 0; kt < nTiles; ++kt) {
    const int cur = kt & 1;
    const bool pref = (kt + 1 < nTiles);
    if (pref) issue_tile(cur ^ 1, (kt + 1) * BK);   // prev buffer free: barrier below last iter
    if (pref) wait_asynccnt_8(); else wait_asynccnt_0();
    __syncthreads();                                 // all waves' copies for `cur` landed

    const unsigned short* __restrict__ sAp = sA[cur];
    const unsigned short* __restrict__ sBp = sB[cur];
#pragma unroll
    for (int kk = 0; kk < BK; kk += 32) {
      FragU aF[4], bF[2];
#pragma unroll
      for (int sm = 0; sm < 4; ++sm) {
        const unsigned short* p =
            &sAp[(waveM * 64 + sm * 16 + m_lane) * LDT + kk + kb];
        aF[sm].h[0] = *(const bf16x8*)(p);
        aF[sm].h[1] = *(const bf16x8*)(p + 16);
      }
#pragma unroll
      for (int sn = 0; sn < 2; ++sn) {
        const unsigned short* p =
            &sBp[(waveN * 32 + sn * 16 + m_lane) * LDT + kk + kb];
        bF[sn].h[0] = *(const bf16x8*)(p);
        bF[sn].h[1] = *(const bf16x8*)(p + 16);
      }
#pragma unroll
      for (int sm = 0; sm < 4; ++sm)
#pragma unroll
        for (int sn = 0; sn < 2; ++sn)
          acc[sm][sn] = __builtin_amdgcn_wmma_f32_16x16x32_bf16(
              false, aF[sm].v, false, bF[sn].v, (short)0, acc[sm][sn],
              false, false);
    }
    __syncthreads();   // everyone done reading `cur` before it is refilled
  }
#else
  // Fallback: synchronous single-buffer staging through VGPRs.
  for (int kt = 0; kt < nTiles; ++kt) {
    const int k0 = kt * BK;
#pragma unroll
    for (int c = 0; c < 4; ++c) {
      int cid = tid + c * 256;
      int row = cid >> 3;
      int ch  = (cid & 7) << 3;
      *(bf16x8*)&sA[0][row * LDT + ch] =
          *(const bf16x8*)&Ab[(size_t)(bm + row) * IN_F + k0 + ch];
      *(bf16x8*)&sB[0][row * LDT + ch] =
          *(const bf16x8*)&Wb[(size_t)(bn + row) * IN_F + k0 + ch];
      if (k0 + BK < IN_F) {
        __builtin_prefetch(&Ab[(size_t)(bm + row) * IN_F + k0 + BK + ch], 0, 0);
        __builtin_prefetch(&Wb[(size_t)(bn + row) * IN_F + k0 + BK + ch], 0, 0);
      }
    }
    __syncthreads();
#pragma unroll
    for (int kk = 0; kk < BK; kk += 32) {
      FragU aF[4], bF[2];
#pragma unroll
      for (int sm = 0; sm < 4; ++sm) {
        const unsigned short* p =
            &sA[0][(waveM * 64 + sm * 16 + m_lane) * LDT + kk + kb];
        aF[sm].h[0] = *(const bf16x8*)(p);
        aF[sm].h[1] = *(const bf16x8*)(p + 16);
      }
#pragma unroll
      for (int sn = 0; sn < 2; ++sn) {
        const unsigned short* p =
            &sB[0][(waveN * 32 + sn * 16 + m_lane) * LDT + kk + kb];
        bF[sn].h[0] = *(const bf16x8*)(p);
        bF[sn].h[1] = *(const bf16x8*)(p + 16);
      }
#pragma unroll
      for (int sm = 0; sm < 4; ++sm)
#pragma unroll
        for (int sn = 0; sn < 2; ++sn)
          acc[sm][sn] = __builtin_amdgcn_wmma_f32_16x16x32_bf16(
              false, aF[sm].v, false, bF[sn].v, (short)0, acc[sm][sn],
              false, false);
    }
    __syncthreads();
  }
#endif

  // --- epilogue: C/D layout => VGPR j holds M=j (lanes 0-15) / M=j+8 (16-31) ---
  const int rhalf = (lane >> 4) << 3;
#pragma unroll
  for (int sn = 0; sn < 2; ++sn) {
    int gn = bn + waveN * 32 + sn * 16 + m_lane;
    float bv = bias[gn];
#pragma unroll
    for (int sm = 0; sm < 4; ++sm) {
      int gm0 = bm + waveM * 64 + sm * 16 + rhalf;
#pragma unroll
      for (int j = 0; j < 8; ++j) {
        C[(size_t)(gm0 + j) * OUT_F + gn] = acc[sm][sn][j] + bv;
      }
    }
  }
}

// ---------------------------------------------------------------------------
// Host-side launcher
// in order: input [8,2048,2048] f32, leaf_tensor [3,4,13,13] f32,
//           weight [2048,2048] f32, bias [2048] f32
// out: [8,2048,2048] f32
// ws: xb (bf16, 64 MiB) | Wb (bf16, 8 MiB)
// ---------------------------------------------------------------------------
extern "C" void kernel_launch(void* const* d_in, const int* in_sizes, int n_in,
                              void* d_out, int out_size, void* d_ws, size_t ws_size,
                              hipStream_t stream) {
  const float* x      = (const float*)d_in[0];
  const float* leaf   = (const float*)d_in[1];
  const float* weight = (const float*)d_in[2];
  const float* bias   = (const float*)d_in[3];
  float* out = (float*)d_out;

  unsigned short* xb = (unsigned short*)d_ws;
  unsigned short* Wb =
      (unsigned short*)((char*)d_ws + (size_t)MTOT * IN_F * sizeof(unsigned short));

  // 1) W + kron-delta -> bf16
  build_w_kernel<<<(OUT_F * IN_F) / 256, 256, 0, stream>>>(leaf, weight, Wb);
  // 2) activations f32 -> bf16
  cvt_x_kernel<<<((size_t)MTOT * IN_F / 4) / 256, 256, 0, stream>>>(x, xb);
  // 3) WMMA GEMM + bias (async double-buffered LDS)
  dim3 grid(OUT_F / BN, MTOT / BM);
  gemm_kernel<<<grid, 256, 0, stream>>>(xb, Wb, bias, out);
}
